// random_69191923138671
// MI455X (gfx1250) — matrix operand order repaired and checked by reading, not compiled, not run on hardware
//
#include <hip/hip_runtime.h>
#include <hip/hip_bf16.h>

// ---- problem constants (from reference) ----
#define SX   96
#define NX   48
#define CIN  512
#define SY   96
#define NYI  64     // items per y-set
#define HSZ  64     // head size
#define NH   2
#define C1   128    // W1 output channels = NH*HSZ

typedef __attribute__((ext_vector_type(16))) _Float16 v16h;
typedef __attribute__((ext_vector_type(8)))  float    v8f;

// ---------------------------------------------------------------------------
// K1: gallery = gelu_exact(x @ W_proj)   [SX*NX, 64] f32
// ---------------------------------------------------------------------------
__global__ void __launch_bounds__(256)
k_gallery(const float* __restrict__ x, const float* __restrict__ Wp,
          float* __restrict__ gal) {
    int idx = blockIdx.x * blockDim.x + threadIdx.x;   // SX*NX*64 threads
    int col = idx & 63;
    int row = idx >> 6;
    const float* xr = x + (size_t)row * CIN;
    float acc = 0.f;
    #pragma unroll 8
    for (int k = 0; k < CIN; ++k)
        acc += xr[k] * Wp[k * 64 + col];
    // exact GELU: v * 0.5 * (1 + erf(v/sqrt(2)))
    gal[idx] = 0.5f * acc * (1.0f + erff(acc * 0.70710678118654752f));
}

// ---------------------------------------------------------------------------
// WMMA operand k-maps (CDNA5 ISA §7.12.2, 16-bit layouts)
// A 16x32: lanes 0-15 -> M, k-group (lane>>4): v0..v3 = K 2j,2j+1 (+grp*8),
//          v4..v7 = K 16+2j,2j+1 (+grp*8)
// B 32x16: lane&15 -> N, K same as A (a_kmap)
// ---------------------------------------------------------------------------
__device__ __forceinline__ int a_kmap(int grp, int half) {
    int vg = half >> 1, pos = half & 1;
    return ((vg >= 4) ? 16 : 0) + ((vg & 3) * 2 + pos) + grp * 8;
}

// ---------------------------------------------------------------------------
// K2: pyPack[y][h][mtile(4)][kstep(2)][lane(32)][half(16)] f16
//     = 0.125 * (y @ W1)   (1/sqrt(HEAD_SIZE) folded in before LeakyReLU)
// ---------------------------------------------------------------------------
__global__ void __launch_bounds__(256)
k_pack_py(const float* __restrict__ y, const float* __restrict__ W1,
          _Float16* __restrict__ pyp) {
    int idx = blockIdx.x * blockDim.x + threadIdx.x;   // 96*2*4*2*32*16 = 786432
    int half  = idx & 15;
    int lane  = (idx >> 4)  & 31;
    int kstep = (idx >> 9)  & 1;
    int mtile = (idx >> 10) & 3;
    int h     = (idx >> 12) & 1;
    int yb    =  idx >> 13;
    int m = mtile * 16 + (lane & 15);
    int k = kstep * 32 + a_kmap(lane >> 4, half);
    int c = h * HSZ + k;
    const float* yr = y + ((size_t)yb * NYI + m) * HSZ;
    float acc = 0.f;
    #pragma unroll
    for (int e = 0; e < HSZ; ++e)
        acc += yr[e] * W1[e * C1 + c];
    pyp[idx] = (_Float16)(acc * 0.125f);
}

// ---------------------------------------------------------------------------
// K3: pxPack[x][h][ntile(3)][kstep(2)][lane(32)][half(16)] f16 = gallery @ W1
// ---------------------------------------------------------------------------
__global__ void __launch_bounds__(256)
k_pack_px(const float* __restrict__ gal, const float* __restrict__ W1,
          _Float16* __restrict__ pxp) {
    int idx = blockIdx.x * blockDim.x + threadIdx.x;   // 96*2*3*2*32*16 = 589824
    int half = idx & 15;
    int r    = idx >> 4;
    int lane = r & 31;  r >>= 5;
    int kstep = r & 1;  r >>= 1;
    int ntile = r % 3;  r /= 3;
    int h  = r & 1;
    int xb = r >> 1;
    int n = ntile * 16 + (lane & 15);
    int k = kstep * 32 + a_kmap(lane >> 4, half);   // B-operand k-map (same as A)
    int c = h * HSZ + k;
    const float* gr = gal + ((size_t)xb * NX + n) * HSZ;
    float acc = 0.f;
    #pragma unroll
    for (int e = 0; e < HSZ; ++e)
        acc += gr[e] * W1[e * C1 + c];
    pxp[idx] = (_Float16)acc;
}

// ---------------------------------------------------------------------------
// K4: fused attention einsum + LeakyReLU + reduction + W2 + normalization.
// One wave per (y,x) pair: D(64x48) = Py(64x64) x Px^T per head, 16x16 tiles,
// K=64 as two chained v_wmma_f32_16x16x32_f16. att never hits memory.
// ---------------------------------------------------------------------------
__global__ void __launch_bounds__(256)
k_att(const _Float16* __restrict__ pyp, const _Float16* __restrict__ pxp,
      const float* __restrict__ nItem, const float* __restrict__ W2,
      float* __restrict__ out) {
    const int lane = threadIdx.x & 31;
    const int wave = threadIdx.x >> 5;
    const int yb = blockIdx.x;                 // 0..95
    const int xb = blockIdx.y * 8 + wave;      // 0..95

    float hs[NH];
    #pragma unroll
    for (int h = 0; h < NH; ++h) {
        // hoist the 6 B tiles (px) for this (x,h): 48 VGPRs
        v16h b[3][2];
        #pragma unroll
        for (int nt = 0; nt < 3; ++nt)
            #pragma unroll
            for (int ks = 0; ks < 2; ++ks)
                b[nt][ks] = *(const v16h*)(pxp +
                    ((((((size_t)xb * NH + h) * 3 + nt) * 2 + ks) * 32) + lane) * 16);

        float s = 0.f;
        #pragma unroll
        for (int mt = 0; mt < 4; ++mt) {
            v16h a0 = *(const v16h*)(pyp +
                ((((((size_t)yb * NH + h) * 4 + mt) * 2 + 0) * 32) + lane) * 16);
            v16h a1 = *(const v16h*)(pyp +
                ((((((size_t)yb * NH + h) * 4 + mt) * 2 + 1) * 32) + lane) * 16);
            #pragma unroll
            for (int nt = 0; nt < 3; ++nt) {
                v8f c = {};
                c = __builtin_amdgcn_wmma_f32_16x16x32_f16(
                        false, a0, false, b[nt][0], (short)0, c, false, false);
                c = __builtin_amdgcn_wmma_f32_16x16x32_f16(
                        false, a1, false, b[nt][1], (short)0, c, false, false);
                // scale (1/8 per operand pair) already folded into py.
                // LeakyReLU(0.3) + accumulate; each lane owns 8 distinct
                // att elements, tiles are disjoint -> wave covers all 64x48.
                #pragma unroll
                for (int j = 0; j < 8; ++j) {
                    float v = c[j];
                    s += (v > 0.f) ? v : 0.3f * v;
                }
            }
        }
        hs[h] = s;
    }

    // wave32 cross-lane reduction
    #pragma unroll
    for (int off = 16; off >= 1; off >>= 1) {
        hs[0] += __shfl_xor(hs[0], off, 32);
        hs[1] += __shfl_xor(hs[1], off, 32);
    }

    if (lane == 0) {
        float inv = 1.0f / (nItem[xb] * (float)NYI);
        out[(size_t)yb * SX + xb] = (hs[0] * W2[0] + hs[1] * W2[1]) * inv;
    }
}

// ---------------------------------------------------------------------------
extern "C" void kernel_launch(void* const* d_in, const int* in_sizes, int n_in,
                              void* d_out, int out_size, void* d_ws, size_t ws_size,
                              hipStream_t stream) {
    const float* x     = (const float*)d_in[0];   // [96,48,512]
    const float* y     = (const float*)d_in[1];   // [96,64,64]
    const float* nItem = (const float*)d_in[2];   // [96]
    const float* Wp    = (const float*)d_in[3];   // [512,64]
    const float* W1    = (const float*)d_in[4];   // [64,128]
    const float* W2    = (const float*)d_in[5];   // [2,1]
    float* out = (float*)d_out;                   // [96,96,1]

    char* ws = (char*)d_ws;
    const size_t galBytes = (size_t)SX * NX * 64 * sizeof(float);      // 1,179,648
    const size_t pypBytes = (size_t)SY * NH * 4 * 2 * 32 * 16 * 2;     // 1,572,864
    float*    gal = (float*)ws;
    _Float16* pyp = (_Float16*)(ws + galBytes);
    _Float16* pxp = (_Float16*)(ws + galBytes + pypBytes);

    k_gallery<<<(SX * NX * 64) / 256, 256, 0, stream>>>(x, Wp, gal);
    k_pack_py<<<(SY * NH * 4 * 2 * 32 * 16) / 256, 256, 0, stream>>>(y, W1, pyp);
    k_pack_px<<<(SX * NH * 3 * 2 * 32 * 16) / 256, 256, 0, stream>>>(gal, W1, pxp);
    dim3 g(SY, SX / 8);
    k_att<<<g, 256, 0, stream>>>(pyp, pxp, nItem, W2, out);
}
